// Transducer_12257836662927
// MI455X (gfx1250) — compile-verified, matrix operand order checked by reading, not run on hardware
//
#include <hip/hip_runtime.h>

// Transducer joint network for MI455X (gfx1250, wave32, WMMA + TDM).
//   * hf = f@W1[:H], hg = g@W1[H:] via f16 WMMA GEMM (f32 accumulate).
//   * Fused big GEMM (M=B*T*U=131072, K=512, N=1024): each block builds a
//     128x512 f16 h-tile (fast_tanh(hf+hg+b1)) in LDS once, and the Tensor
//     Data Mover streams 64-column chunks of W2^T into double-buffered LDS
//     (pad_enable inserts the 16B/row pad that makes ds_load_b128 fragment
//     reads bank-conflict-free). All WMMA operands come from LDS.
//   * Output write (512MB f32 ~= 23us at 23.3 TB/s) is the roofline floor;
//     wave tile 32x32 (mt=2 x nt=2 per wave) keeps fragment traffic at
//     16 FLOP/byte so LDS bandwidth doesn't dominate.

typedef __attribute__((ext_vector_type(16))) _Float16 v16h;
typedef __attribute__((ext_vector_type(8)))  _Float16 v8h;
typedef __attribute__((ext_vector_type(4)))  _Float16 v4h;
typedef __attribute__((ext_vector_type(8)))  float    v8f;
typedef __attribute__((ext_vector_type(4)))  float    v4f;
typedef __attribute__((ext_vector_type(4)))  unsigned v4u;
typedef __attribute__((ext_vector_type(8)))  unsigned v8u;

#define LDS_STRIDE 520           // 512 + 8 halfwords: 4-dword bank shift/row
#define H_TILE_ELEMS (128 * LDS_STRIDE)          // 66560 f16 = 133120 B
#define B_BUF_ELEMS  (64 * LDS_STRIDE)           // 33280 f16 =  66560 B

// 16-bit WMMA fragment: lane(0..15) row/col = lane&15; elements 0..7 = K+0..7,
// 8..15 = K+16..23; lanes 16..31 take +8/+24, folded in via (lane>>4)*8.
__device__ __forceinline__ v16h load_frag16(const _Float16* p) {
  v8h lo = *(const v8h*)p;
  v8h hi = *(const v8h*)(p + 16);
  return __builtin_shufflevector(lo, hi, 0, 1, 2, 3, 4, 5, 6, 7,
                                         8, 9, 10, 11, 12, 13, 14, 15);
}

#define WMMA_F32_F16(a, b, c)                                              \
  __builtin_amdgcn_wmma_f32_16x16x32_f16(false, (a), false, (b), (short)0, \
                                         (c), false, false)

// ------------------------------------------------------------- TDM --------
// 1-D tensor_load_to_lds of 16384 dwords (64 cols x 512 f16 of W2^T) with
// LDS padding: 4 dwords inserted every 256 dwords -> stride 520 halfwords.
// Descriptor per CDNA5 ISA ch.8: group0 = {flags, lds_addr, global_addr,
// type=2}; group1 = {data_size=4B, pad_enable, pad_interval=256dw(code 7),
// pad_amount=4dw(code 3), tensor_dim0=tile_dim0=16384}.
__device__ __forceinline__ void tdm_load_chunk(const _Float16* gsrc,
                                               unsigned lds_byte_addr) {
  unsigned long long ga = (unsigned long long)(uintptr_t)gsrc;
  v4u g0;
  g0.x = 1u;                                            // count=1, user D#
  g0.y = lds_byte_addr;                                 // lds_addr
  g0.z = (unsigned)ga;                                  // global_addr[31:0]
  g0.w = (unsigned)((ga >> 32) & 0x1FFFFFFu) | (2u << 30);  // addr hi | type=2
  v8u g1;
  g1[0] = (2u << 16)      // data_size = 4B
        | (1u << 20)      // pad_enable
        | (7u << 22)      // pad_interval code 7 = 256 dwords
        | (3u << 25);     // pad_amount  code 3 = 4 dwords (16B)
  g1[1] = (16384u & 0xFFFFu) << 16;      // tensor_dim0[15:0]
  g1[2] = (16384u >> 16) | (1u << 16);   // tensor_dim0[31:16] | tensor_dim1=1
  g1[3] = 16384u << 16;                  // tile_dim0 = 16384 dwords
  g1[4] = 0u;                            // tile_dim1/2 unused
  g1[5] = 16384u;                        // tensor_dim0_stride
  g1[6] = 0u;
  g1[7] = 0u;
  asm volatile("tensor_load_to_lds %0, %1" : : "s"(g0), "s"(g1) : "memory");
}

// ---------------------------------------------------------------- prep ----
__global__ void cvt_f32_f16(const float* __restrict__ src,
                            _Float16* __restrict__ dst, int n) {
  int i = blockIdx.x * blockDim.x + threadIdx.x;
  if (i < n) dst[i] = (_Float16)src[i];
}

__global__ void transpose_f32_f16(const float* __restrict__ src,
                                  _Float16* __restrict__ dst,
                                  int rows, int cols) {
  int i = blockIdx.x * blockDim.x + threadIdx.x;
  if (i < rows * cols) {
    int r = i / cols;
    int c = i - r * cols;
    dst[(size_t)c * rows + r] = (_Float16)src[i];
  }
}

// --------------------------------------------- generic WMMA GEMM ----------
// C[MxN] f32 row-major = A[MxK] f16 row-major * Bt[NxK]^T f16.
// Block tile 128x64 (8 waves, wave tile 16x64); used for the small hf/hg
// projections where operands are tiny and L2-hot.
__global__ __launch_bounds__(256) void wmma_gemm_rrt(
    const _Float16* __restrict__ A, const _Float16* __restrict__ Bt,
    float* __restrict__ C, int N, int K) {
  const int lane  = threadIdx.x & 31;
  const int wave  = threadIdx.x >> 5;
  const int l16   = lane & 15;
  const int hi    = lane >> 4;
  const int mbase = blockIdx.x * 128 + wave * 16;
  const int nbase = blockIdx.y * 64;

  v8f acc[4] = {};
  const _Float16* aP = A + (size_t)(mbase + l16) * K + hi * 8;

  for (int k0 = 0; k0 < K; k0 += 32) {
    v16h a = load_frag16(aP + k0);
#pragma unroll
    for (int j = 0; j < 4; ++j) {
      const _Float16* bP = Bt + (size_t)(nbase + j * 16 + l16) * K + hi * 8 + k0;
      v16h b = load_frag16(bP);
      acc[j] = WMMA_F32_F16(a, b, acc[j]);
    }
  }
#pragma unroll
  for (int j = 0; j < 4; ++j) {
    int col = nbase + j * 16 + l16;
#pragma unroll
    for (int r = 0; r < 8; ++r)
      C[(size_t)(mbase + hi * 8 + r) * N + col] = acc[j][r];
  }
}

// ------------------------------------------------- fused joint GEMM -------
// LDS: [ h-tile 128x520 f16 | Bbuf0 64x520 f16 | Bbuf1 64x520 f16 ] = 260KB.
// 8 waves = 4 M-strips (32 rows) x 2 N-groups (32 cols); 16 N-chunks of 64.
__global__ __launch_bounds__(256) void joint_wmma_kernel(
    const float* __restrict__ hf, const float* __restrict__ hg,
    const float* __restrict__ b1, const _Float16* __restrict__ W2t,
    const float* __restrict__ b2, float* __restrict__ out) {
  extern __shared__ _Float16 sh[];
  const int tid   = threadIdx.x;
  const int wave  = tid >> 5;
  const int mBase = blockIdx.x * 128;
  // Low 32 bits of a flat shared-aperture address are the LDS byte offset
  // (ISA 10.2 aperture mapping) -> uniform, lives in SGPRs for the D#.
  const unsigned ldsB = (unsigned)(uintptr_t)(void*)(sh + H_TILE_ELEMS);

  // Kick off DMA of W2^T chunk 0 while we build the h tile.
  if (wave == 0) tdm_load_chunk(W2t, ldsB);

  // Phase 1: h tile, vectorized; each h element computed exactly once.
  for (int e4 = tid * 4; e4 < 128 * 512; e4 += 256 * 4) {
    int row  = e4 >> 9;
    int k    = e4 & 511;
    int m    = mBase + row;                      // flat (b,t,u)
    int rowf = m >> 7;                           // m / U
    int rowg = ((m >> 15) << 7) | (m & 127);     // b*U + u
    v4f xf = *(const v4f*)&hf[(size_t)rowf * 512 + k];
    v4f xg = *(const v4f*)&hg[(size_t)rowg * 512 + k];
    v4f xb = *(const v4f*)&b1[k];
    v4h o;
#pragma unroll
    for (int i = 0; i < 4; ++i) {
      float s = xf[i] + xg[i] + xb[i];
      o[i] = (_Float16)(s / (1.0f + __builtin_fabsf(s)));  // fast_tanh
    }
    *(v4h*)&sh[row * LDS_STRIDE + k] = o;
  }
  if (wave == 0) __builtin_amdgcn_s_wait_tensorcnt(0);
  __syncthreads();  // h tile + B chunk 0 visible to all waves

  const int lane   = tid & 31;
  const int l16    = lane & 15;
  const int hi     = lane >> 4;
  const int mstrip = wave & 3;    // 4 strips of 32 rows
  const int ngrp   = wave >> 2;   // 2 groups of 32 cols

  const _Float16* aB0 = &sh[(mstrip * 32 + l16) * LDS_STRIDE + hi * 8];
  const _Float16* aB1 = aB0 + 16 * LDS_STRIDE;

  for (int nc = 0; nc < 16; ++nc) {
    const int cur = nc & 1;
    // Prefetch next chunk into the other buffer; overlaps this chunk's math.
    if (wave == 0 && nc + 1 < 16)
      tdm_load_chunk(W2t + (size_t)(nc + 1) * 64 * 512,
                     ldsB + (unsigned)(1 - cur) * (B_BUF_ELEMS * 2));

    const _Float16* bB0 = sh + H_TILE_ELEMS + cur * B_BUF_ELEMS +
                          (ngrp * 32 + l16) * LDS_STRIDE + hi * 8;
    const _Float16* bB1 = bB0 + 16 * LDS_STRIDE;

    v8f acc00 = {}, acc01 = {}, acc10 = {}, acc11 = {};
    for (int k0 = 0; k0 < 512; k0 += 32) {
      v16h a0 = load_frag16(aB0 + k0);          // all frags via ds_load_b128
      v16h a1 = load_frag16(aB1 + k0);
      v16h b0 = load_frag16(bB0 + k0);
      v16h b1v = load_frag16(bB1 + k0);
      acc00 = WMMA_F32_F16(a0, b0, acc00);
      acc01 = WMMA_F32_F16(a0, b1v, acc01);
      acc10 = WMMA_F32_F16(a1, b0, acc10);
      acc11 = WMMA_F32_F16(a1, b1v, acc11);
    }

    const int colB = nc * 64 + ngrp * 32;
    const float bias0 = b2[colB + l16];
    const float bias1 = b2[colB + 16 + l16];
    const int row0 = mBase + mstrip * 32 + hi * 8;
#pragma unroll
    for (int r = 0; r < 8; ++r) {
      out[(size_t)(row0 + r) * 1024 + colB + l16]           = acc00[r] + bias0;
      out[(size_t)(row0 + r) * 1024 + colB + 16 + l16]      = acc01[r] + bias1;
      out[(size_t)(row0 + 16 + r) * 1024 + colB + l16]      = acc10[r] + bias0;
      out[(size_t)(row0 + 16 + r) * 1024 + colB + 16 + l16] = acc11[r] + bias1;
    }

    // Next chunk's DMA must land; barrier also fences buffer reuse.
    if (wave == 0) __builtin_amdgcn_s_wait_tensorcnt(0);
    __syncthreads();
  }
}

// ------------------------------------------------------------- launch -----
extern "C" void kernel_launch(void* const* d_in, const int* in_sizes, int n_in,
                              void* d_out, int out_size, void* d_ws,
                              size_t ws_size, hipStream_t stream) {
  (void)in_sizes; (void)n_in; (void)out_size; (void)ws_size;
  // B=4 T=256 U=128 H=512 P=512 HID=512 V=1024
  const float* f  = (const float*)d_in[0];
  const float* g  = (const float*)d_in[1];
  const float* W1 = (const float*)d_in[2];
  const float* b1 = (const float*)d_in[3];
  const float* W2 = (const float*)d_in[4];
  const float* b2 = (const float*)d_in[5];
  float* out = (float*)d_out;

  char* ws = (char*)d_ws;                       // ~6.5 MB total
  float*    hf   = (float*)(ws + 0);            // 1024*512 f32
  float*    hg   = (float*)(ws + 2097152);      //  512*512 f32
  _Float16* f16f = (_Float16*)(ws + 3145728);   // 1024*512 f16
  _Float16* g16  = (_Float16*)(ws + 4194304);   //  512*512 f16
  _Float16* W1at = (_Float16*)(ws + 4718592);   // (HID x H)^T f16
  _Float16* W1bt = (_Float16*)(ws + 5242880);   // (HID x P)^T f16
  _Float16* W2t  = (_Float16*)(ws + 5767168);   // (V x HID)^T f16

  const int nf = 4 * 256 * 512;
  const int ng = 4 * 128 * 512;
  cvt_f32_f16<<<(nf + 255) / 256, 256, 0, stream>>>(f, f16f, nf);
  cvt_f32_f16<<<(ng + 255) / 256, 256, 0, stream>>>(g, g16, ng);
  transpose_f32_f16<<<(512 * 512 + 255) / 256, 256, 0, stream>>>(
      W1, W1at, 512, 512);
  transpose_f32_f16<<<(512 * 512 + 255) / 256, 256, 0, stream>>>(
      W1 + 512 * 512, W1bt, 512, 512);
  transpose_f32_f16<<<(512 * 1024 + 255) / 256, 256, 0, stream>>>(
      W2, W2t, 512, 1024);

  dim3 gf(1024 / 128, 512 / 64);
  wmma_gemm_rrt<<<gf, 256, 0, stream>>>(f16f, W1at, hf, 512, 512);
  dim3 gg(512 / 128, 512 / 64);
  wmma_gemm_rrt<<<gg, 256, 0, stream>>>(g16, W1bt, hg, 512, 512);

  dim3 gj(131072 / 128);
  size_t shbytes = (size_t)(H_TILE_ELEMS + 2 * B_BUF_ELEMS) * sizeof(_Float16);
  joint_wmma_kernel<<<gj, 256, shbytes, stream>>>(hf, hg, b1, W2t, b2, out);
}
